// TransformerBlock_58471684767954
// MI455X (gfx1250) — matrix operand order repaired and checked by reading, not compiled, not run on hardware
//
#include <hip/hip_runtime.h>

typedef __attribute__((ext_vector_type(16))) _Float16 v16h;
typedef __attribute__((ext_vector_type(8)))  _Float16 v8h;
typedef __attribute__((ext_vector_type(8)))  float    v8f;

#define WMMA_F16(a, b, c) \
  __builtin_amdgcn_wmma_f32_16x16x32_f16(false, (a), false, (b), (short)0, (c), false, false)

// ---------------------------------------------------------------------------
// Fragment loaders (f16, K-major source rows).
// A 16x32 frag: lane r holds row r; contiguous K = {kh..kh+15}, kh=16*(lane>=16)
__device__ __forceinline__ v16h load_fragA(const _Float16* base, int row0, int k0,
                                           int ld, int lane) {
  const int r  = lane & 15;
  const int kh = (lane >> 4) << 4;
  const _Float16* p = base + (size_t)(row0 + r) * ld + k0 + kh;
  return *(const v16h*)p;
}

// B 32x16 frag: lane r holds column r (= row r of the [N,K] weight matrix);
// contiguous K-run of 16 starting at kb = 16*(lane>=16)  -> one 32B load.
__device__ __forceinline__ v16h load_fragB(const _Float16* base, int col0, int k0,
                                           int ld, int lane) {
  const int r  = lane & 15;
  const int kb = (lane >> 4) << 4;
  const _Float16* p = base + (size_t)(col0 + r) * ld + k0 + kb;
  return *(const v16h*)p;
}

// ---------------------------------------------------------------------------
__global__ __launch_bounds__(256) void cvt_kernel(const float* __restrict__ in,
                                                  _Float16* __restrict__ out, int n) {
  int i = blockIdx.x * 256 + threadIdx.x;
  if (i < n) out[i] = (_Float16)in[i];
}

// ---------------------------------------------------------------------------
// Y[M,N] = X[M,K] @ W[N,K]^T + bias, 64x64 tile per 4-wave workgroup.
// MODE 0: f16 out   MODE 1: f32 out   MODE 2: f32 exact-GELU out
template <int MODE>
__global__ __launch_bounds__(128)
void gemm_bias_kernel(const _Float16* __restrict__ X, const _Float16* __restrict__ W,
                      const float* __restrict__ bias,
                      _Float16* __restrict__ Yh, float* __restrict__ Yf,
                      int M, int N, int K) {
  const int lane = threadIdx.x & 31;
  const int wave = threadIdx.x >> 5;
  const int tilesN = N >> 6;
  const int tm = blockIdx.x / tilesN;
  const int tn = blockIdx.x - tm * tilesN;
  const int row0 = (tm << 6) + (wave << 4);
  const int col0 = tn << 6;

  v8f c[4] = {};
  for (int k0 = 0; k0 < K; k0 += 32) {
    v16h a = load_fragA(X, row0, k0, K, lane);
#pragma unroll
    for (int j = 0; j < 4; ++j) {
      v16h bw = load_fragB(W, col0 + j * 16, k0, K, lane);
      c[j] = WMMA_F16(a, bw, c[j]);
    }
  }

  const int n  = lane & 15;
  const int mo = (lane >> 4) << 3;
#pragma unroll
  for (int j = 0; j < 4; ++j) {
    const int col = col0 + j * 16 + n;
    const float bj = bias[col];
#pragma unroll
    for (int i = 0; i < 8; ++i) {
      const float y = c[j][i] + bj;
      const size_t idx = (size_t)(row0 + i + mo) * N + col;
      if (MODE == 0) {
        Yh[idx] = (_Float16)y;
      } else if (MODE == 1) {
        Yf[idx] = y;
      } else {
        Yf[idx] = 0.5f * y * (1.0f + erff(y * 0.7071067811865475f));
      }
    }
  }
}

// ---------------------------------------------------------------------------
// Fused flash attention for one (b, h, 64-row q tile). Each of 4 waves owns 16
// q rows. Per 32-key tile: scores via 4 WMMA, online softmax via 16-lane
// shuffles, P transposed through LDS into A-frag layout, V tile transposed
// into LDS for B-frags, P@V via 4 WMMA. Head (b,h) is a contiguous 1024x64
// block (pure reshape); output scattered to [b, l, h*64 + d].
__global__ __launch_bounds__(128)
void attention_kernel(const _Float16* __restrict__ qH, const _Float16* __restrict__ kH,
                      const _Float16* __restrict__ vH, const int* __restrict__ mask,
                      _Float16* __restrict__ atth) {
  __shared__ __align__(32) _Float16 p_lds[4][16 * 32];
  __shared__ __align__(32) _Float16 vt[64 * 32];

  const int lane = threadIdx.x & 31;
  const int wave = threadIdx.x >> 5;
  const int qt = blockIdx.x & 15;
  const int h  = (blockIdx.x >> 4) & 15;
  const int b  = blockIdx.x >> 8;
  const size_t headoff = (size_t)b * 1048576 + (size_t)h * 65536;
  const _Float16* qb = qH + headoff;
  const _Float16* kb = kH + headoff;
  const _Float16* vb = vH + headoff;
  const int* mb = mask + b * 1024;
  const int qrow0 = qt * 64 + wave * 16;

  const v16h a0 = load_fragA(qb, qrow0,  0, 64, lane);
  const v16h a1 = load_fragA(qb, qrow0, 32, 64, lane);

  v8f o[4] = {};
  float mrow[8], lrow[8];
#pragma unroll
  for (int i = 0; i < 8; ++i) { mrow[i] = -3.0e38f; lrow[i] = 0.0f; }
  const int n  = lane & 15;
  const int mo = (lane >> 4) << 3;
  _Float16* pw = p_lds[wave];

  for (int k0 = 0; k0 < 1024; k0 += 32) {
    __syncthreads();  // protect vt reads of previous iteration
    for (int t = threadIdx.x; t < 2048; t += 128) {
      const int kk = t >> 6;
      const int d  = t & 63;
      vt[d * 32 + kk] = vb[(size_t)(k0 + kk) * 64 + d];  // coalesced read
    }
    __syncthreads();

    // ---- scores: 16(q) x 32(k), K = DH = 64 chained over two WMMAs
    v8f s0 = {}, s1 = {};
    {
      v16h bk;
      bk = load_fragB(kb, k0,       0, 64, lane); s0 = WMMA_F16(a0, bk, s0);
      bk = load_fragB(kb, k0,      32, 64, lane); s0 = WMMA_F16(a1, bk, s0);
      bk = load_fragB(kb, k0 + 16,  0, 64, lane); s1 = WMMA_F16(a0, bk, s1);
      bk = load_fragB(kb, k0 + 16, 32, 64, lane); s1 = WMMA_F16(a1, bk, s1);
    }

    const bool ok0 = mb[k0 + n] != 0;
    const bool ok1 = mb[k0 + 16 + n] != 0;
#pragma unroll
    for (int i = 0; i < 8; ++i) {
      const float e0 = ok0 ? s0[i] * 0.125f : -1.0e10f;
      const float e1 = ok1 ? s1[i] * 0.125f : -1.0e10f;
      // row max across the 16-lane half-wave (one row per VGPR index)
      float rm = fmaxf(e0, e1);
      rm = fmaxf(rm, __shfl_xor(rm, 1, 32));
      rm = fmaxf(rm, __shfl_xor(rm, 2, 32));
      rm = fmaxf(rm, __shfl_xor(rm, 4, 32));
      rm = fmaxf(rm, __shfl_xor(rm, 8, 32));
      const float mnew = fmaxf(mrow[i], rm);
      const float corr = __expf(mrow[i] - mnew);
      mrow[i] = mnew;
      const float p0 = __expf(e0 - mnew);
      const float p1 = __expf(e1 - mnew);
      float rs = p0 + p1;
      rs += __shfl_xor(rs, 1, 32);
      rs += __shfl_xor(rs, 2, 32);
      rs += __shfl_xor(rs, 4, 32);
      rs += __shfl_xor(rs, 8, 32);
      lrow[i] = lrow[i] * corr + rs;
      o[0][i] *= corr; o[1][i] *= corr; o[2][i] *= corr; o[3][i] *= corr;
      // stash P (C layout) to LDS so it can be re-read in A-frag layout
      pw[(i + mo) * 32 + n]      = (_Float16)p0;
      pw[(i + mo) * 32 + 16 + n] = (_Float16)p1;
    }

    // ---- reload P as A-fragment (same wave; compiler inserts s_wait_dscnt)
    v16h pa;
    {
      const int r   = lane & 15;
      const int kh2 = (lane >> 4) << 4;
      const _Float16* pr = pw + r * 32;
#pragma unroll
      for (int t = 0; t < 16; ++t) { pa[t] = pr[kh2 + t]; }
    }

    // ---- P(16x32) @ V(32x64): B-frag column d = row d of transposed vt
#pragma unroll
    for (int j = 0; j < 4; ++j) {
      const int r   = lane & 15;
      const int kb2 = (lane >> 4) << 4;
      const v16h bv = *(const v16h*)&vt[(j * 16 + r) * 32 + kb2];
      o[j] = WMMA_F16(pa, bv, o[j]);
    }
  }

  // epilogue: normalize and scatter to [b, l, h*64 + d]
#pragma unroll
  for (int i = 0; i < 8; ++i) {
    const float inv = 1.0f / lrow[i];
    const int l = qrow0 + i + mo;
#pragma unroll
    for (int j = 0; j < 4; ++j) {
      const int d = j * 16 + n;
      atth[(size_t)b * 1048576 + (size_t)l * 1024 + h * 64 + d] = (_Float16)(o[j][i] * inv);
    }
  }
}

// ---------------------------------------------------------------------------
// skip = LayerNorm(A + B) * g + be ; writes f32 (+ optional f16 copy)
__global__ __launch_bounds__(256)
void add_ln_kernel(const float* __restrict__ A, const float* __restrict__ Bv,
                   const float* __restrict__ g, const float* __restrict__ be,
                   float* __restrict__ outF, _Float16* __restrict__ outH) {
  __shared__ float red[256];
  __shared__ float red2[256];
  const int t = threadIdx.x;
  const size_t base = (size_t)blockIdx.x * 1024;
  float v[4];
  float s = 0.f, s2 = 0.f;
#pragma unroll
  for (int i = 0; i < 4; ++i) {
    const float x = A[base + t + 256 * i] + Bv[base + t + 256 * i];
    v[i] = x; s += x; s2 += x * x;
  }
  red[t] = s; red2[t] = s2;
  __syncthreads();
  for (int off = 128; off > 0; off >>= 1) {
    if (t < off) { red[t] += red[t + off]; red2[t] += red2[t + off]; }
    __syncthreads();
  }
  const float mean = red[0] * (1.0f / 1024.0f);
  const float var  = red2[0] * (1.0f / 1024.0f) - mean * mean;
  const float rstd = rsqrtf(var + 1e-5f);
#pragma unroll
  for (int i = 0; i < 4; ++i) {
    const int e = t + 256 * i;
    const float y = (v[i] - mean) * rstd * g[e] + be[e];
    outF[base + e] = y;
    if (outH) outH[base + e] = (_Float16)y;
  }
}

// out = x*alpha + LayerNorm(A + B)*(1-alpha)
__global__ __launch_bounds__(256)
void add_ln_blend_kernel(const float* __restrict__ A, const float* __restrict__ Bv,
                         const float* __restrict__ g, const float* __restrict__ be,
                         const float* __restrict__ x, const float* __restrict__ alpha_p,
                         float* __restrict__ out) {
  __shared__ float red[256];
  __shared__ float red2[256];
  const int t = threadIdx.x;
  const size_t base = (size_t)blockIdx.x * 1024;
  float v[4];
  float s = 0.f, s2 = 0.f;
#pragma unroll
  for (int i = 0; i < 4; ++i) {
    const float xv = A[base + t + 256 * i] + Bv[base + t + 256 * i];
    v[i] = xv; s += xv; s2 += xv * xv;
  }
  red[t] = s; red2[t] = s2;
  __syncthreads();
  for (int off = 128; off > 0; off >>= 1) {
    if (t < off) { red[t] += red[t + off]; red2[t] += red2[t + off]; }
    __syncthreads();
  }
  const float mean = red[0] * (1.0f / 1024.0f);
  const float var  = red2[0] * (1.0f / 1024.0f) - mean * mean;
  const float rstd = rsqrtf(var + 1e-5f);
  const float alpha = alpha_p[0];
#pragma unroll
  for (int i = 0; i < 4; ++i) {
    const int e = t + 256 * i;
    const float y = (v[i] - mean) * rstd * g[e] + be[e];
    out[base + e] = x[base + e] * alpha + y * (1.0f - alpha);
  }
}

// ---------------------------------------------------------------------------
extern "C" void kernel_launch(void* const* d_in, const int* in_sizes, int n_in,
                              void* d_out, int out_size, void* d_ws, size_t ws_size,
                              hipStream_t stream) {
  (void)in_sizes; (void)n_in; (void)out_size; (void)ws_size;
  const float* x    = (const float*)d_in[0];
  const int*   mask = (const int*)d_in[1];
  const float* Wq = (const float*)d_in[2];
  const float* bq = (const float*)d_in[3];
  const float* Wk = (const float*)d_in[4];
  const float* bk = (const float*)d_in[5];
  const float* Wv = (const float*)d_in[6];
  const float* bv = (const float*)d_in[7];
  const float* Wo = (const float*)d_in[8];
  const float* bo = (const float*)d_in[9];
  const float* g1 = (const float*)d_in[10];
  const float* b1 = (const float*)d_in[11];
  const float* Wf = (const float*)d_in[12];
  const float* bf = (const float*)d_in[13];
  const float* g2 = (const float*)d_in[14];
  const float* b2 = (const float*)d_in[15];
  const float* alpha = (const float*)d_in[16];

  const int Mr = 4096;          // B*L
  const int E  = 1024;
  const size_t nAct = (size_t)Mr * E;   // 4,194,304
  const size_t nW   = (size_t)E * E;

  char* ws = (char*)d_ws;
  size_t off = 0;
  auto take = [&](size_t bytes) -> char* {
    char* p = ws + off;
    off += (bytes + 255) & ~(size_t)255;
    return p;
  };
  _Float16* xh     = (_Float16*)take(nAct * 2);
  _Float16* wqh    = (_Float16*)take(nW * 2);
  _Float16* wkh    = (_Float16*)take(nW * 2);
  _Float16* wvh    = (_Float16*)take(nW * 2);
  _Float16* woh    = (_Float16*)take(nW * 2);
  _Float16* wfh    = (_Float16*)take(nW * 2);
  _Float16* qh     = (_Float16*)take(nAct * 2);
  _Float16* khB    = (_Float16*)take(nAct * 2);
  _Float16* vhB    = (_Float16*)take(nAct * 2);
  _Float16* atth   = (_Float16*)take(nAct * 2);
  float*    atto   = (float*)take(nAct * 4);
  float*    skip1  = (float*)take(nAct * 4);
  _Float16* skip1h = (_Float16*)take(nAct * 2);
  float*    fcc    = (float*)take(nAct * 4);

  auto cvt = [&](const float* src, _Float16* dst, int n) {
    cvt_kernel<<<dim3((n + 255) / 256), dim3(256), 0, stream>>>(src, dst, n);
  };
  cvt(x, xh, (int)nAct);
  cvt(Wq, wqh, (int)nW);
  cvt(Wk, wkh, (int)nW);
  cvt(Wv, wvh, (int)nW);
  cvt(Wo, woh, (int)nW);
  cvt(Wf, wfh, (int)nW);

  const dim3 gG((Mr / 64) * (E / 64));  // 1024 workgroups
  const dim3 gB(128);
  gemm_bias_kernel<0><<<gG, gB, 0, stream>>>(xh, wqh, bq, qh,  nullptr, Mr, E, E);
  gemm_bias_kernel<0><<<gG, gB, 0, stream>>>(xh, wkh, bk, khB, nullptr, Mr, E, E);
  gemm_bias_kernel<0><<<gG, gB, 0, stream>>>(xh, wvh, bv, vhB, nullptr, Mr, E, E);

  attention_kernel<<<dim3(1024), dim3(128), 0, stream>>>(qh, khB, vhB, mask, atth);

  gemm_bias_kernel<1><<<gG, gB, 0, stream>>>(atth, woh, bo, nullptr, atto, Mr, E, E);

  add_ln_kernel<<<dim3(Mr), dim3(256), 0, stream>>>(x, atto, g1, b1, skip1, skip1h);

  gemm_bias_kernel<2><<<gG, gB, 0, stream>>>(skip1h, wfh, bf, nullptr, fcc, Mr, E, E);

  add_ln_blend_kernel<<<dim3(Mr), dim3(256), 0, stream>>>(skip1, fcc, g2, b2, x, alpha,
                                                          (float*)d_out);
}
